// ParticleNet_57397942944343
// MI455X (gfx1250) — matrix operand order repaired and checked by reading, not compiled, not run on hardware
//
#include <hip/hip_runtime.h>
#include <hip/hip_bf16.h>

typedef __attribute__((ext_vector_type(16))) _Float16 v16h;
typedef __attribute__((ext_vector_type(8)))  _Float16 h8;
typedef __attribute__((ext_vector_type(8)))  float    v8f;

#define NB 256
#define NP 128
#define NPts (NB * NP)
#define KNN 16
#define NE (NPts * KNN)
#define EPSV 1e-5f
#define MLP_ROWS 64   // edges per block (4 waves x 16-row M-tiles)

__device__ __forceinline__ float gelu_f(float v) {
    return 0.5f * v * (1.0f + erff(v * 0.70710678118654752f));
}
__device__ __forceinline__ float psi_f(float p) {
    return copysignf(log1pf(fabsf(p)), p);
}

// ---------------------------------------------------------------------------
// Weight transpose + fp32 -> f16 convert:  WT[n*d + k] = W[k*d + n]
// ---------------------------------------------------------------------------
__global__ void convert_wt_kernel(const float* __restrict__ W,
                                  _Float16* __restrict__ WT, int d) {
    int idx = blockIdx.x * blockDim.x + threadIdx.x;
    if (idx < d * d) {
        int n = idx / d, k = idx - n * d;
        WT[idx] = (_Float16)W[k * d + n];
    }
}

__global__ void zero_stats_kernel(float* stats) {
    if (threadIdx.x < 8) stats[threadIdx.x] = 0.0f;
}

// ---------------------------------------------------------------------------
// KNN (per batch) + edge features (norms, dots) + global feature statistics
// One block per batch; one thread per point.
// ---------------------------------------------------------------------------
__global__ __launch_bounds__(NP) void knn_feats_kernel(
    const float* __restrict__ x, float* __restrict__ feats,
    float* __restrict__ xdiff, float* __restrict__ stats) {
    __shared__ float4 xs[NP];
    __shared__ float red[5];
    const int t = threadIdx.x;
    const int b = blockIdx.x;
    const float4 me = ((const float4*)x)[b * NP + t];
    xs[t] = me;
    if (t < 5) red[t] = 0.0f;
    __syncthreads();

    float best[KNN];
    int   bidx[KNN];
#pragma unroll
    for (int r = 0; r < KNN; ++r) { best[r] = 3.0e38f; bidx[r] = 0; }

    for (int j = 0; j < NP; ++j) {
        float4 o = xs[j];
        float d0 = me.x - o.x, d1 = me.y - o.y, d2 = me.z - o.z, d3 = me.w - o.w;
        float dd = d0 * d0 + d1 * d1 + d2 * d2 + d3 * d3;
        if (j == t) dd += 1.0e10f;
        if (dd < best[KNN - 1]) {
            int p = KNN - 1;
            while (p > 0 && best[p - 1] > dd) {
                best[p] = best[p - 1]; bidx[p] = bidx[p - 1]; --p;
            }
            best[p] = dd; bidx[p] = j;
        }
    }

    float s0 = 0.f, s1 = 0.f, s00 = 0.f, s11 = 0.f, s01 = 0.f;
    const int i = b * NP + t;
#pragma unroll 4
    for (int r = 0; r < KNN; ++r) {
        float4 o = xs[bidx[r]];
        float d0 = me.x - o.x, d1 = me.y - o.y, d2 = me.z - o.z, d3 = me.w - o.w;
        // Minkowski metric (+,-,-,-)
        float n2 = d0 * d0 - d1 * d1 - d2 * d2 - d3 * d3;
        float dt = me.x * o.x - me.y * o.y - me.z * o.z - me.w * o.w;
        float fn = psi_f(n2), fd = psi_f(dt);
        int e = i * KNN + r;
        feats[2 * e + 0] = fn;
        feats[2 * e + 1] = fd;
        ((float4*)xdiff)[e] = make_float4(d0, d1, d2, d3);
        s0 += fn; s1 += fd; s00 += fn * fn; s11 += fd * fd; s01 += fn * fd;
    }
    atomicAdd(&red[0], s0);
    atomicAdd(&red[1], s1);
    atomicAdd(&red[2], s00);
    atomicAdd(&red[3], s11);
    atomicAdd(&red[4], s01);
    __syncthreads();
    if (t < 5) atomicAdd(&stats[t], red[t]);
}

// ---------------------------------------------------------------------------
// Fold bn1 -> lin1 -> bn2 into per-channel affine map of (f0, f1):
//   pre-gelu_c = alpha_c*f0 + beta_c*f1 + gamma_c
// bn2 statistics computed analytically from the 2x2 feature covariance.
// ---------------------------------------------------------------------------
__global__ void prep_abg_kernel(const float* __restrict__ stats,
                                const float* __restrict__ bn1w,
                                const float* __restrict__ bn1b,
                                const float* __restrict__ lin1w,
                                const float* __restrict__ bn2w,
                                const float* __restrict__ bn2b,
                                float* __restrict__ abg, int d) {
    const float invE = 1.0f / (float)NE;
    float m0 = stats[0] * invE, m1 = stats[1] * invE;
    float v0 = stats[2] * invE - m0 * m0;
    float v1 = stats[3] * invE - m1 * m1;
    float cv = stats[4] * invE - m0 * m1;
    float s0 = rsqrtf(v0 + EPSV), s1 = rsqrtf(v1 + EPSV);
    float a0 = s0 * bn1w[0], a1 = s1 * bn1w[1];
    float c0 = bn1b[0] - m0 * a0, c1 = bn1b[1] - m1 * a1;
    for (int c = threadIdx.x; c < d; c += blockDim.x) {
        float L0 = lin1w[c], L1 = lin1w[d + c];
        float A = a0 * L0, B = a1 * L1, C = c0 * L0 + c1 * L1;
        float mu  = A * m0 + B * m1 + C;
        float var = A * A * v0 + B * B * v1 + 2.0f * A * B * cv;
        float tsc = rsqrtf(var + EPSV) * bn2w[c];
        abg[c]         = A * tsc;
        abg[d + c]     = B * tsc;
        abg[2 * d + c] = C * tsc + bn2b[c] - mu * tsc;
    }
}

// ---------------------------------------------------------------------------
// Fused edge MLP with WMMA f16->f32 (16x16x32):
//   g  = gelu(affine(feats))            [64 x D]  (LDS, f16)
//   h2 = gelu(g @ W2 + b2)              [64 x D]  (WMMA, LDS f16)
//   h3 = h2 * sigmoid(h2 . m_w + m_b)   [64 x D]  (LDS f16)
//   scale = gelu(h3 @ X1 + x1b) . x2w   [64]      (WMMA + butterfly reduce)
// 128 threads (4 waves), 16-edge M-tile per wave, weights streamed from L2
// as pre-transposed f16 (one contiguous 32B fragment per lane). Dynamic LDS
// (2 x 64 x (D+8) halves) so D=256 fits gfx1250's 320KB/WG budget.
// ---------------------------------------------------------------------------
template <int D>
__global__ __launch_bounds__(128) void edge_mlp_kernel(
    const float* __restrict__ feats, const float* __restrict__ abg,
    const _Float16* __restrict__ W2T, const float* __restrict__ b2,
    const float* __restrict__ mw, const float* __restrict__ mb,
    const _Float16* __restrict__ X1T, const float* __restrict__ x1b,
    const float* __restrict__ x2w, float* __restrict__ scaleOut) {
    constexpr int STRIDE = D + 8;  // halves per row (+16B pad -> 4-bank shift)
    extern __shared__ float4 smem4[];
    _Float16* bufA = (_Float16*)smem4;
    _Float16* bufB = bufA + MLP_ROWS * STRIDE;

    const int lane  = threadIdx.x & 31;
    const int wave  = threadIdx.x >> 5;       // 0..3
    const int rlow  = lane & 15;              // row within wave tile / N column
    const int hi    = (lane >> 4) & 1;        // half-wave selector
    const int wrow  = wave * 16;              // first row of this wave
    const int eBase = blockIdx.x * MLP_ROWS;  // first edge of this block

    if (lane == 0) {
        __builtin_prefetch(W2T + wave * 2048, 0, 1);
        __builtin_prefetch(X1T + wave * 2048, 0, 1);
    }

    // ---- stage 1: g = gelu(alpha*f0 + beta*f1 + gamma) -> bufA (f16) ----
    {
        const int row = wrow + rlow;
        const int e = eBase + row;
        const float f0 = feats[2 * e + 0];
        const float f1 = feats[2 * e + 1];
        const int c0 = hi ? (D / 2) : 0;
        for (int c = c0; c < c0 + D / 2; c += 8) {
            h8 o;
#pragma unroll
            for (int u = 0; u < 8; ++u) {
                float t = abg[c + u] * f0 + abg[D + c + u] * f1 + abg[2 * D + c + u];
                o[u] = (_Float16)gelu_f(t);
            }
            *(h8*)&bufA[row * STRIDE + c] = o;
        }
    }
    __syncthreads();

    // ---- stage 2: h2 = gelu(g @ W2 + b2) -> bufB (WMMA) ----
    for (int nt = 0; nt < D / 16; ++nt) {
        v8f acc = {};
#pragma unroll
        for (int kk = 0; kk < D / 32; ++kk) {
            // A fragment (16-bit A 16x32 layout)
            const _Float16* arow = &bufA[(wrow + rlow) * STRIDE + kk * 32 + hi * 16];
            v16h a;
            reinterpret_cast<float4*>(&a)[0] = *reinterpret_cast<const float4*>(arow);
            reinterpret_cast<float4*>(&a)[1] = *reinterpret_cast<const float4*>(arow + 8);
            // B fragment: column nt*16+rlow, K = kk*32 + hi*16 .. +15 (contig)
            const _Float16* brow = &W2T[(nt * 16 + rlow) * D + kk * 32 + hi * 16];
            v16h bf = *reinterpret_cast<const v16h*>(brow);
            acc = __builtin_amdgcn_wmma_f32_16x16x32_f16(
                false, a, false, bf, (short)0, acc, false, false);
        }
        const int col = nt * 16 + rlow;
        const float bias = b2[col];
#pragma unroll
        for (int r = 0; r < 8; ++r) {
            const int m = r + hi * 8;
            bufB[(wrow + m) * STRIDE + col] = (_Float16)gelu_f(acc[r] + bias);
        }
    }
    __syncthreads();

    // ---- stage 3: gate = sigmoid(h2 . m_w + m_b); h3 = h2*gate -> bufA ----
    {
        const int row = wrow + rlow;
        const int c0 = hi ? (D / 2) : 0;
        float part = 0.f;
        for (int c = c0; c < c0 + D / 2; c += 8) {
            h8 v = *(const h8*)&bufB[row * STRIDE + c];
            float4 w0 = *(const float4*)&mw[c];
            float4 w1 = *(const float4*)&mw[c + 4];
            part += (float)v[0] * w0.x + (float)v[1] * w0.y +
                    (float)v[2] * w0.z + (float)v[3] * w0.w +
                    (float)v[4] * w1.x + (float)v[5] * w1.y +
                    (float)v[6] * w1.z + (float)v[7] * w1.w;
        }
        float tot = part + __shfl_xor(part, 16, 32);
        float gate = 1.0f / (1.0f + __expf(-(tot + mb[0])));
        for (int c = c0; c < c0 + D / 2; c += 8) {
            h8 v = *(const h8*)&bufB[row * STRIDE + c];
            h8 o;
#pragma unroll
            for (int u = 0; u < 8; ++u)
                o[u] = (_Float16)((float)v[u] * gate);
            *(h8*)&bufA[row * STRIDE + c] = o;
        }
    }
    __syncthreads();

    // ---- stage 4: scale = gelu(h3 @ X1 + x1b) . x2w (WMMA + reduce) ----
    float part[8];
#pragma unroll
    for (int r = 0; r < 8; ++r) part[r] = 0.f;

    for (int nt = 0; nt < D / 16; ++nt) {
        v8f acc = {};
#pragma unroll
        for (int kk = 0; kk < D / 32; ++kk) {
            const _Float16* arow = &bufA[(wrow + rlow) * STRIDE + kk * 32 + hi * 16];
            v16h a;
            reinterpret_cast<float4*>(&a)[0] = *reinterpret_cast<const float4*>(arow);
            reinterpret_cast<float4*>(&a)[1] = *reinterpret_cast<const float4*>(arow + 8);
            const _Float16* brow = &X1T[(nt * 16 + rlow) * D + kk * 32 + hi * 16];
            v16h bf = *reinterpret_cast<const v16h*>(brow);
            acc = __builtin_amdgcn_wmma_f32_16x16x32_f16(
                false, a, false, bf, (short)0, acc, false, false);
        }
        const int col = nt * 16 + rlow;
        const float bias = x1b[col];
        const float w2 = x2w[col];
#pragma unroll
        for (int r = 0; r < 8; ++r)
            part[r] += gelu_f(acc[r] + bias) * w2;
    }
    // butterfly reduce within each 16-lane half (C layout: N = lane&15)
#pragma unroll
    for (int r = 0; r < 8; ++r) {
        float v = part[r];
        v += __shfl_xor(v, 1, 32);
        v += __shfl_xor(v, 2, 32);
        v += __shfl_xor(v, 4, 32);
        v += __shfl_xor(v, 8, 32);
        part[r] = v;
    }
    if (rlow == 0) {
#pragma unroll
        for (int r = 0; r < 8; ++r)
            scaleOut[eBase + wrow + hi * 8 + r] = part[r];
    }
}

// ---------------------------------------------------------------------------
// x_out = x_in + mean_k(scale * x_diff)
// ---------------------------------------------------------------------------
__global__ __launch_bounds__(256) void aggregate_kernel(
    const float* __restrict__ xin, const float* __restrict__ scale,
    const float* __restrict__ xdiff, float* __restrict__ xout) {
    int i = blockIdx.x * blockDim.x + threadIdx.x;
    if (i >= NPts) return;
    float4 a = make_float4(0.f, 0.f, 0.f, 0.f);
    const float4* xd = (const float4*)xdiff;
#pragma unroll
    for (int r = 0; r < KNN; ++r) {
        int e = i * KNN + r;
        float s = scale[e];
        float4 d = xd[e];
        a.x += s * d.x; a.y += s * d.y; a.z += s * d.z; a.w += s * d.w;
    }
    const float ik = 1.0f / (float)KNN;
    float4 xi = ((const float4*)xin)[i];
    ((float4*)xout)[i] =
        make_float4(xi.x + a.x * ik, xi.y + a.y * ik, xi.z + a.z * ik, xi.w + a.w * ik);
}

// ---------------------------------------------------------------------------
// Host orchestration
// ---------------------------------------------------------------------------
extern "C" void kernel_launch(void* const* d_in, const int* in_sizes, int n_in,
                              void* d_out, int out_size, void* d_ws, size_t ws_size,
                              hipStream_t stream) {
    (void)in_sizes; (void)n_in; (void)out_size; (void)ws_size;
    static const int DIMS_[3] = {64, 128, 256};

    const float* x_in = (const float*)d_in[0];

    // workspace carve-out (all offsets 256B aligned)
    char* ws = (char*)d_ws;
    size_t off = 0;
    auto take = [&](size_t bytes) {
        char* p = ws + off;
        off += (bytes + 255) & ~size_t(255);
        return p;
    };
    float* x0    = (float*)take((size_t)NPts * 4 * sizeof(float));
    float* x1    = (float*)take((size_t)NPts * 4 * sizeof(float));
    float* feats = (float*)take((size_t)NE * 2 * sizeof(float));
    float* xdiff = (float*)take((size_t)NE * 4 * sizeof(float));
    float* scale = (float*)take((size_t)NE * sizeof(float));
    float* stats = (float*)take(8 * sizeof(float));
    float* abg   = (float*)take(3 * 256 * sizeof(float));
    _Float16* W2T = (_Float16*)take((size_t)256 * 256 * sizeof(_Float16));
    _Float16* X1T = (_Float16*)take((size_t)256 * 256 * sizeof(_Float16));

    hipMemcpyAsync(x0, x_in, (size_t)NPts * 4 * sizeof(float),
                   hipMemcpyDeviceToDevice, stream);

    const float* cur = x0;
    for (int layer = 0; layer < 3; ++layer) {
        const int d = DIMS_[layer];
        const int base = 3 + layer * 12;
        const float* bn1w  = (const float*)d_in[base + 0];
        const float* bn1b  = (const float*)d_in[base + 1];
        const float* lin1w = (const float*)d_in[base + 2];
        const float* bn2w  = (const float*)d_in[base + 3];
        const float* bn2b  = (const float*)d_in[base + 4];
        const float* lin2w = (const float*)d_in[base + 5];
        const float* lin2b = (const float*)d_in[base + 6];
        const float* m_w   = (const float*)d_in[base + 7];
        const float* m_b   = (const float*)d_in[base + 8];
        const float* x1w   = (const float*)d_in[base + 9];
        const float* x1b   = (const float*)d_in[base + 10];
        const float* x2w   = (const float*)d_in[base + 11];

        const int wgrid = (d * d + 255) / 256;
        convert_wt_kernel<<<wgrid, 256, 0, stream>>>(lin2w, W2T, d);
        convert_wt_kernel<<<wgrid, 256, 0, stream>>>(x1w, X1T, d);
        zero_stats_kernel<<<1, 8, 0, stream>>>(stats);

        knn_feats_kernel<<<NB, NP, 0, stream>>>(cur, feats, xdiff, stats);
        prep_abg_kernel<<<1, 256, 0, stream>>>(stats, bn1w, bn1b, lin1w,
                                               bn2w, bn2b, abg, d);

        const int mgrid = NE / MLP_ROWS;
        const size_t shmem = (size_t)2 * MLP_ROWS * (d + 8) * sizeof(_Float16);
        switch (d) {
        case 64:
            edge_mlp_kernel<64><<<mgrid, 128, shmem, stream>>>(
                feats, abg, W2T, lin2b, m_w, m_b, X1T, x1b, x2w, scale);
            break;
        case 128:
            edge_mlp_kernel<128><<<mgrid, 128, shmem, stream>>>(
                feats, abg, W2T, lin2b, m_w, m_b, X1T, x1b, x2w, scale);
            break;
        default:
            edge_mlp_kernel<256><<<mgrid, 128, shmem, stream>>>(
                feats, abg, W2T, lin2b, m_w, m_b, X1T, x1b, x2w, scale);
            break;
        }

        float* out = (layer == 2) ? (float*)d_out : ((cur == x0) ? x1 : x0);
        aggregate_kernel<<<NPts / 256, 256, 0, stream>>>(cur, scale, xdiff, out);
        cur = out;
    }
}
